// PointnetSetAbstraction_33449205301354
// MI455X (gfx1250) — matrix operand order repaired and checked by reading, not compiled, not run on hardware
//
#include <hip/hip_runtime.h>
#include <hip/hip_bf16.h>
#include <stdint.h>

// ---------------- problem constants (match reference setup_inputs) ----------
#define B_    16
#define N_    4096
#define C_    128
#define P_    1024
#define S_    32
#define RAD2  0.16f           // 0.4^2
#define EPS_  1e-5f
#define MTOT  ((long)B_ * P_ * S_)   // 524288 GEMM columns

typedef __bf16 bf16;
typedef __attribute__((ext_vector_type(16))) __bf16 v16bf;
typedef __attribute__((ext_vector_type(8)))  float  v8f;

// ---------------- bf16 helpers ----------------------------------------------
__device__ inline unsigned short f2bf(float f) {
  unsigned u = __float_as_uint(f);
  u += 0x7fffu + ((u >> 16) & 1u);          // round-to-nearest-even
  return (unsigned short)(u >> 16);
}
__device__ inline float bf2f(unsigned short h) {
  return __uint_as_float(((unsigned)h) << 16);
}

// WMMA-B tile layout, contiguous k order (32x16 KxN bf16 per 16x16x32 step):
//   lanes 0-15 hold K=0..15 (col = lane), lanes 16-31 hold K=16..31,
//   element j within a lane's 16-bf16 slice = k & 15.
// Tile (nt, kt) occupies 1024 bytes (512 ushorts) at ((nt*KSTEPS)+kt)*512.
// => for a fixed column m, channels kt*32+half*16+(0..15) are CONTIGUOUS
//    32-byte runs at ushort offset ((half ? col+16 : col) * 16).

// ---------------- weight packing (A-matrix order used here) ---------------------
// A matrix 16x32 (MxK) bf16, contiguous k order (matches B packing):
//   lanes 0-15 : M=0..15, elements 0-15 -> K 0..15
//   lanes 16-31: M=0..15, elements 0-15 -> K 16..31
__global__ void pack_w(const float* __restrict__ w, unsigned short* __restrict__ Wp,
                       int Cout, int Cin, int ksteps, int total) {
  int gid = blockIdx.x * blockDim.x + threadIdx.x;
  if (gid >= total) return;                 // total = (Cout/16)*ksteps*32
  int tile = gid >> 5;
  int lane = gid & 31;
  int ot = tile / ksteps;
  int kt = tile % ksteps;
  int row = lane & 15;
  int hi  = lane >> 4;
  union { unsigned short s[16]; uint4 q[2]; } t;
#pragma unroll
  for (int j = 0; j < 16; ++j) {
    int k = (hi ? 16 : 0) + j;
    int kg = kt * 32 + k;
    float val = (kg < Cin) ? w[(size_t)(ot * 16 + row) * Cin + kg] : 0.0f;
    t.s[j] = f2bf(val);
  }
  uint4* dst = (uint4*)(Wp + (size_t)tile * 512);
  dst[lane * 2 + 0] = t.q[0];
  dst[lane * 2 + 1] = t.q[1];
}

// ---------------- ball query + group + pack into X0 (B128 stores) -----------
__global__ void group_pack(const float* __restrict__ xyz,
                           const float* __restrict__ feat,
                           float* __restrict__ out_xyz,
                           int*   __restrict__ out_idx,
                           unsigned short* __restrict__ X0) {
  __shared__ float sx[N_ * 3];              // 48 KB of the 320 KB WGP LDS
  const int b   = blockIdx.y;
  const int tid = threadIdx.x;
  for (int i = tid; i < N_ * 3; i += blockDim.x)
    sx[i] = xyz[(size_t)b * N_ * 3 + i];
  __syncthreads();

  const int p = blockIdx.x * blockDim.x + tid;   // p < P_
  const float c0 = sx[p * 3 + 0], c1 = sx[p * 3 + 1], c2 = sx[p * 3 + 2];
  out_xyz[((size_t)b * P_ + p) * 3 + 0] = c0;
  out_xyz[((size_t)b * P_ + p) * 3 + 1] = c1;
  out_xyz[((size_t)b * P_ + p) * 3 + 2] = c2;
  out_idx[(size_t)b * P_ + p] = p;

  int idxs[S_];
  int cnt = 0;
  for (int j = 0; j < N_; ++j) {
    float dx = sx[j * 3 + 0] - c0;
    float dy = sx[j * 3 + 1] - c1;
    float dz = sx[j * 3 + 2] - c2;
    if (dx * dx + dy * dy + dz * dz < RAD2) {
      idxs[cnt++] = j;
      if (cnt == S_) break;
    }
  }
  if (cnt == 0) idxs[cnt++] = p;            // query point is always in-radius
  for (int s = cnt; s < S_; ++s) idxs[s] = idxs[0];  // pad with first hit

  const long mbase = ((long)b * P_ + p) * S_;
  for (int s = 0; s < S_; ++s) {
    int j = idxs[s];
    long m   = mbase + s;
    long nt  = m >> 4;
    int  col = (int)(m & 15);
    float rel[3] = { sx[j*3+0]-c0, sx[j*3+1]-c1, sx[j*3+2]-c2 };
    for (int kt = 0; kt < 5; ++kt) {
      union { unsigned short s16[32]; uint4 q[4]; } t;
#pragma unroll
      for (int jj = 0; jj < 32; ++jj) {
        int k = kt * 32 + jj;
        float val;
        if (k < 3)            val = rel[k];
        else if (k < C_ + 3)  val = feat[((size_t)b * C_ + (k - 3)) * N_ + j];
        else                  val = 0.0f;
        t.s16[jj] = f2bf(val);
      }
      uint4* dst = (uint4*)(X0 + (nt * 5 + kt) * 512);
      dst[col * 2 + 0]        = t.q[0];     // K  0..15 of this kt
      dst[col * 2 + 1]        = t.q[1];
      dst[(col + 16) * 2 + 0] = t.q[2];     // K 16..31 of this kt
      dst[(col + 16) * 2 + 1] = t.q[3];
    }
  }
}

// ---------------- WMMA GEMM --------------------------------------------------
// Block tile: 64 out x 64 m.  4 waves: wave w owns out-rows [ot=by*4+w]*16,
// reuses its A tile across 4 WMMAs (m sub-tiles).  The four B tiles per k-step
// are staged ONCE in block-shared LDS, double-buffered via async DMA
// (global_load_async_to_lds_b128 / ASYNCcnt); wave w DMAs sub-tile w.
template <int KSTEPS>
__global__ __launch_bounds__(128) void gemm_bf16(
    const unsigned short* __restrict__ Wp, const unsigned short* __restrict__ Xp,
    unsigned short* __restrict__ raw, float* __restrict__ sum,
    float* __restrict__ sumsq, long Mtot) {
  __shared__ uint4 sB[2][4][64];            // [buf][m sub-tile][1KB]
  const int lane = threadIdx.x & 31;
  const int wv   = threadIdx.x >> 5;
  const int ot   = blockIdx.y * 4 + wv;     // 16-row output tile index
  const long nt0 = (long)blockIdx.x * 4;    // first of four 16-col m tiles
  const unsigned long long xbase = (unsigned long long)(uintptr_t)Xp;

  auto issue = [&](int kt, int buf) {       // wave wv stages B tile (nt0+wv,kt)
    unsigned lb = (unsigned)(uintptr_t)(&sB[buf][wv][0]) + (unsigned)lane * 16u;
    unsigned go = (unsigned)(((nt0 + wv) * KSTEPS + kt) * 1024) +
                  (unsigned)lane * 16u;
    asm volatile("global_load_async_to_lds_b128 %0, %1, %2"
                 :: "v"(lb), "v"(go), "s"(xbase) : "memory");
    asm volatile("global_load_async_to_lds_b128 %0, %1, %2"
                 :: "v"(lb + 512u), "v"(go + 512u), "s"(xbase) : "memory");
  };

  issue(0, 0);
  v8f acc0 = {}, acc1 = {}, acc2 = {}, acc3 = {};
  for (int kt = 0; kt < KSTEPS; ++kt) {
    const int buf = kt & 1;
    asm volatile("s_wait_asynccnt 0" ::: "memory");  // own DMAs for buf done
    __syncthreads();          // all waves' DMAs done; all reads of buf^1 done
    if (kt + 1 < KSTEPS) {
      __builtin_prefetch(Wp + ((size_t)ot * KSTEPS + kt + 1) * 512, 0, 3);
      issue(kt + 1, buf ^ 1); // safe: nobody reads buf^1's old data anymore
    }
    union { uint4 q[2]; v16bf v; } ua, ub0, ub1, ub2, ub3;
    const uint4* wq = (const uint4*)(Wp + ((size_t)ot * KSTEPS + kt) * 512);
    ua.q[0] = wq[lane * 2 + 0];
    ua.q[1] = wq[lane * 2 + 1];
    ub0.q[0] = sB[buf][0][lane * 2]; ub0.q[1] = sB[buf][0][lane * 2 + 1];
    ub1.q[0] = sB[buf][1][lane * 2]; ub1.q[1] = sB[buf][1][lane * 2 + 1];
    ub2.q[0] = sB[buf][2][lane * 2]; ub2.q[1] = sB[buf][2][lane * 2 + 1];
    ub3.q[0] = sB[buf][3][lane * 2]; ub3.q[1] = sB[buf][3][lane * 2 + 1];
    acc0 = __builtin_amdgcn_wmma_f32_16x16x32_bf16(false, ua.v, false, ub0.v,
                                                   (short)0, acc0, false, false);
    acc1 = __builtin_amdgcn_wmma_f32_16x16x32_bf16(false, ua.v, false, ub1.v,
                                                   (short)0, acc1, false, false);
    acc2 = __builtin_amdgcn_wmma_f32_16x16x32_bf16(false, ua.v, false, ub2.v,
                                                   (short)0, acc2, false, false);
    acc3 = __builtin_amdgcn_wmma_f32_16x16x32_bf16(false, ua.v, false, ub3.v,
                                                   (short)0, acc3, false, false);
  }

  // D layout: reg r, lanes 0-15 -> row r, lanes 16-31 -> row r+8; col = lane&15
  const int colsel = lane & 15;
  const int hi8 = (lane >> 4) << 3;
#pragma unroll
  for (int r = 0; r < 8; ++r) {
    int ch = ot * 16 + hi8 + r;
    float v0 = acc0[r], v1 = acc1[r], v2 = acc2[r], v3 = acc3[r];
    raw[(size_t)ch * Mtot + (nt0 + 0) * 16 + colsel] = f2bf(v0);
    raw[(size_t)ch * Mtot + (nt0 + 1) * 16 + colsel] = f2bf(v1);
    raw[(size_t)ch * Mtot + (nt0 + 2) * 16 + colsel] = f2bf(v2);
    raw[(size_t)ch * Mtot + (nt0 + 3) * 16 + colsel] = f2bf(v3);
    float s1 = v0 + v1 + v2 + v3;
    float s2 = v0 * v0 + v1 * v1 + v2 * v2 + v3 * v3;
#pragma unroll
    for (int off = 1; off < 16; off <<= 1) {
      s1 += __shfl_xor(s1, off, 32);
      s2 += __shfl_xor(s2, off, 32);
    }
    if (colsel == 0) {
      atomicAdd(&sum[ch], s1);
      atomicAdd(&sumsq[ch], s2);
    }
  }
}

// ---------------- BN stats -> per-channel scale/shift ------------------------
__global__ void finalize_stats(const float* __restrict__ sum,
                               const float* __restrict__ sumsq,
                               const float* __restrict__ g,
                               const float* __restrict__ beta,
                               float* __restrict__ scale,
                               float* __restrict__ shift, int C, float invM) {
  int c = blockIdx.x * blockDim.x + threadIdx.x;
  if (c >= C) return;
  float m  = sum[c] * invM;
  float vr = sumsq[c] * invM - m * m;
  float a  = g[c] * rsqrtf(vr + EPS_);
  scale[c] = a;
  shift[c] = beta[c] - m * a;               // bias b cancels inside BN
}

// ---------------- fused BN+ReLU applied while repacking next layer input ----
// One thread per column m; emits WMMA-B tiles (contiguous k) with B128 stores.
__global__ void norm_relu_pack(const unsigned short* __restrict__ raw,
                               const float* __restrict__ scale,
                               const float* __restrict__ shift,
                               unsigned short* __restrict__ Xn, long Mtot) {
  long m = (long)blockIdx.x * blockDim.x + threadIdx.x;
  if (m >= Mtot) return;
  long nt  = m >> 4;
  int  col = (int)(m & 15);
  for (int kt = 0; kt < 4; ++kt) {          // C = 128 channels, 4 k-steps
    union { unsigned short s16[32]; uint4 q[4]; } t;
#pragma unroll
    for (int jj = 0; jj < 32; ++jj) {
      int c = kt * 32 + jj;
      float x = bf2f(raw[(size_t)c * Mtot + m]);
      t.s16[jj] = f2bf(fmaxf(scale[c] * x + shift[c], 0.0f));
    }
    uint4* dst = (uint4*)(Xn + (nt * 4 + kt) * 512);
    dst[col * 2 + 0]        = t.q[0];
    dst[col * 2 + 1]        = t.q[1];
    dst[(col + 16) * 2 + 0] = t.q[2];
    dst[(col + 16) * 2 + 1] = t.q[3];
  }
}

// ---------------- final BN+ReLU + max over the 32 samples --------------------
__global__ void norm_relu_max(const unsigned short* __restrict__ raw,
                              const float* __restrict__ scale,
                              const float* __restrict__ shift,
                              float* __restrict__ outF) {
  long gid = (long)blockIdx.x * blockDim.x + threadIdx.x;
  if (gid >= (long)B_ * 256 * P_) return;
  int p = (int)(gid % P_);
  int o = (int)((gid / P_) % 256);
  int b = (int)(gid / ((long)256 * P_));
  long mb = ((long)b * P_ + p) * S_;
  float a = scale[o], d = shift[o];
  float mx = 0.0f;                          // ReLU output is >= 0
  for (int s = 0; s < S_; ++s) {
    float x = bf2f(raw[(size_t)o * MTOT + mb + s]);
    mx = fmaxf(mx, fmaxf(a * x + d, 0.0f));
  }
  outF[((size_t)b * 256 + o) * P_ + p] = mx;
}

// ---------------- host launcher ---------------------------------------------
extern "C" void kernel_launch(void* const* d_in, const int* in_sizes, int n_in,
                              void* d_out, int out_size, void* d_ws, size_t ws_size,
                              hipStream_t stream) {
  (void)in_sizes; (void)n_in; (void)out_size;
  const float* xyz   = (const float*)d_in[0];
  const float* feat  = (const float*)d_in[1];
  const float* w0    = (const float*)d_in[3];
  const float* g0    = (const float*)d_in[5];
  const float* beta0 = (const float*)d_in[6];
  const float* w1    = (const float*)d_in[7];
  const float* g1    = (const float*)d_in[9];
  const float* beta1 = (const float*)d_in[10];
  const float* w2    = (const float*)d_in[11];
  const float* g2    = (const float*)d_in[13];
  const float* beta2 = (const float*)d_in[14];

  float* outF    = (float*)d_out;
  float* out_xyz = outF;                                   // B*P*3
  float* out_nf  = outF + (size_t)B_ * P_ * 3;             // B*256*P
  int*   out_idx = (int*)(outF + (size_t)B_ * P_ * 3 + (size_t)B_ * 256 * P_);

  // workspace carving (requires ~440 MB)
  char* ws = (char*)d_ws;
  size_t off = 0;
  auto carve = [&](size_t bytes) { size_t r = off; off = (off + bytes + 255) & ~(size_t)255; return r; };
  unsigned short* Wp0 = (unsigned short*)(ws + carve(8 * 5 * 1024));
  unsigned short* Wp1 = (unsigned short*)(ws + carve(8 * 4 * 1024));
  unsigned short* Wp2 = (unsigned short*)(ws + carve(16 * 4 * 1024));
  float* stats = (float*)(ws + carve(3 * 4 * 256 * sizeof(float)));
  unsigned short* Xbuf = (unsigned short*)(ws + carve((size_t)160 * MTOT * 2)); // X0/X1/X2 (reused)
  unsigned short* Rbuf = (unsigned short*)(ws + carve((size_t)256 * MTOT * 2)); // raw0/1/2 (reused)
  (void)ws_size;

  float* sum0 = stats + 0 * 256;  float* sq0 = stats + 1 * 256;
  float* sc0  = stats + 2 * 256;  float* sh0 = stats + 3 * 256;
  float* sum1 = stats + 4 * 256;  float* sq1 = stats + 5 * 256;
  float* sc1  = stats + 6 * 256;  float* sh1 = stats + 7 * 256;
  float* sum2 = stats + 8 * 256;  float* sq2 = stats + 9 * 256;
  float* sc2  = stats + 10 * 256; float* sh2 = stats + 11 * 256;

  const float invM = 1.0f / (float)MTOT;
  hipMemsetAsync(stats, 0, 3 * 4 * 256 * sizeof(float), stream);

  // pack weights into WMMA-A order (K padded to multiples of 32)
  pack_w<<<(8 * 5 * 32 + 127) / 128, 128, 0, stream>>>(w0, Wp0, 128, C_ + 3, 5, 8 * 5 * 32);
  pack_w<<<(8 * 4 * 32 + 127) / 128, 128, 0, stream>>>(w1, Wp1, 128, 128, 4, 8 * 4 * 32);
  pack_w<<<(16 * 4 * 32 + 127) / 128, 128, 0, stream>>>(w2, Wp2, 256, 128, 4, 16 * 4 * 32);

  // ball query + group + pack (also emits new_xyz and sample_idxs)
  group_pack<<<dim3(P_ / 128, B_), 128, 0, stream>>>(xyz, feat, out_xyz, out_idx, Xbuf);

  const int mblk = (int)(MTOT / 64);         // 8192 blocks of 4 m-tiles
  // layer 0: 131(->160) -> 128
  gemm_bf16<5><<<dim3(mblk, 2), 128, 0, stream>>>(Wp0, Xbuf, Rbuf, sum0, sq0, MTOT);
  finalize_stats<<<1, 128, 0, stream>>>(sum0, sq0, g0, beta0, sc0, sh0, 128, invM);
  norm_relu_pack<<<(int)(MTOT / 256), 256, 0, stream>>>(Rbuf, sc0, sh0, Xbuf, MTOT);
  // layer 1: 128 -> 128
  gemm_bf16<4><<<dim3(mblk, 2), 128, 0, stream>>>(Wp1, Xbuf, Rbuf, sum1, sq1, MTOT);
  finalize_stats<<<1, 128, 0, stream>>>(sum1, sq1, g1, beta1, sc1, sh1, 128, invM);
  norm_relu_pack<<<(int)(MTOT / 256), 256, 0, stream>>>(Rbuf, sc1, sh1, Xbuf, MTOT);
  // layer 2: 128 -> 256
  gemm_bf16<4><<<dim3(mblk, 4), 128, 0, stream>>>(Wp2, Xbuf, Rbuf, sum2, sq2, MTOT);
  finalize_stats<<<1, 256, 0, stream>>>(sum2, sq2, g2, beta2, sc2, sh2, 256, invM);
  // fused BN + ReLU + max over samples
  norm_relu_max<<<(int)(((long)B_ * 256 * P_) / 256), 256, 0, stream>>>(Rbuf, sc2, sh2, out_nf);
}
